// StackedHighwayBiLSTM_44367012168045
// MI455X (gfx1250) — matrix operand order repaired and checked by reading, not compiled, not run on hardware
//
#include <hip/hip_runtime.h>
#include <hip/hip_bf16.h>

// ---------------------------------------------------------------------------
// StackedHighwayBiLSTM on MI455X (gfx1250, wave32, WMMA + TDM)
//
// All GEMMs run through v_wmma_f32_16x16x32_f16 (f16 operands, f32 accum).
// The recurrent GEMM stages h (64x512 f16 = 64KB) into LDS once per block via
// the Tensor Data Mover (tensor_load_to_lds + s_wait_tensorcnt), then the
// K-loop reads A fragments from LDS (ds_load_b128) while streaming Whh
// fragments from global (L2-resident). Each wave register-blocks all 4 M
// tiles (B=64) so one B fragment feeds 4 WMMAs.
// ---------------------------------------------------------------------------

typedef __attribute__((ext_vector_type(16))) _Float16 v16h;
typedef __attribute__((ext_vector_type(8)))  _Float16 v8h;
typedef __attribute__((ext_vector_type(8)))  float    v8f;
typedef __attribute__((ext_vector_type(4)))  unsigned int u32x4;
typedef __attribute__((ext_vector_type(8)))  int          i32x8;
typedef __attribute__((ext_vector_type(4)))  int          i32x4;

#define TT   256
#define BB   64
#define HH   512
#define DIN0 512
#define NG   2048   // 4H
#define D2H  1024   // 2H
#define MROWS (TT * BB)   // 16384

#if defined(__AMDGCN__) && __has_builtin(__builtin_amdgcn_tensor_load_to_lds)
#define USE_TDM 1
#else
#define USE_TDM 0
#endif

union FragU { v16h f; v8h h[2]; };

// --- Fragment loaders (ISA 7.12.2 layouts, row-major [rows, K] source) -----

// A matrix 16x32 f16: lanes 0-15 = rows M0..M0+15, v0..3 = K0..7 pairs,
// v4..7 = K16..23; lanes 16-31 same rows, K8..15 / K24..31.
__device__ __forceinline__ v16h load_a_frag(const _Float16* __restrict__ base,
                                            int lda, int m0, int k0, int lane) {
    const int row = m0 + (lane & 15);
    const int kh  = (lane >> 4) * 8;
    const _Float16* p = base + (size_t)row * lda + k0;
    FragU u;
    u.h[0] = *(const v8h*)(p + kh);
    u.h[1] = *(const v8h*)(p + 16 + kh);
    return u.f;
}

// B matrix 32x16 f16 from row-major W[N,K] (computing X @ W^T):
// lane = column; lanes 0-15 hold K=k0..7,k0+16..23, lanes 16-31 K=k0+8..15,k0+24..31.
__device__ __forceinline__ v16h load_b_frag(const _Float16* __restrict__ w,
                                            int ldw, int n0, int k0, int lane) {
    const int n  = n0 + (lane & 15);
    const int kb = k0 + (lane >> 4) * 8;
    const _Float16* p = w + (size_t)n * ldw + kb;
    FragU u;
    u.h[0] = *(const v8h*)(p);
    u.h[1] = *(const v8h*)(p + 16);
    return u.f;
}

__device__ __forceinline__ v8f wmma_f16(v16h a, v16h b, v8f c) {
    return __builtin_amdgcn_wmma_f32_16x16x32_f16(false, a, false, b,
                                                  (short)0, c, false, false);
}

__device__ __forceinline__ float sigmoidf_(float x) {
    return 1.0f / (1.0f + __expf(-x));
}

// --- f32 -> f16 conversion -------------------------------------------------
__global__ __launch_bounds__(256) void f32_to_f16_kernel(
    const float* __restrict__ src, _Float16* __restrict__ dst, int n) {
    int i = blockIdx.x * blockDim.x + threadIdx.x;
    if (i < n) dst[i] = (_Float16)src[i];
}

// --- init h/c state --------------------------------------------------------
__global__ __launch_bounds__(256) void init_state_kernel(
    const float* __restrict__ f_init_l,   // [2,H] : h0, c0
    const float* __restrict__ b_init_l,
    float* __restrict__ c32, float* __restrict__ h32,
    _Float16* __restrict__ h16) {
    const int dir = blockIdx.z;
    const int idx = blockIdx.x * blockDim.x + threadIdx.x;   // 0 .. B*H-1
    const int h   = idx & (HH - 1);
    const float* ini = dir ? b_init_l : f_init_l;
    const size_t st = (size_t)dir * BB * HH + idx;
    const float h0 = ini[h], c0 = ini[HH + h];
    h32[st] = h0; c32[st] = c0; h16[st] = (_Float16)h0;
}

// --- gates_x = X @ Wih^T + b  (both directions via z) ----------------------
// grid (N/16=128, M/(16*8)=128, 2), block 256 (8 waves, wave -> mTile)
__global__ __launch_bounds__(256) void gx_gemm_kernel(
    const _Float16* __restrict__ inp, int K,
    const _Float16* __restrict__ wih_f, const _Float16* __restrict__ wih_b,
    const float* __restrict__ bias_f,   const float* __restrict__ bias_b,
    _Float16* __restrict__ gx_f,        _Float16* __restrict__ gx_b) {
    const int lane = threadIdx.x & 31;
    const int wave = threadIdx.x >> 5;
    const int nT   = blockIdx.x;
    const int mT   = blockIdx.y * 8 + wave;
    const int dir  = blockIdx.z;
    const _Float16* w    = dir ? wih_b  : wih_f;
    const float*    bias = dir ? bias_b : bias_f;
    _Float16*       gx   = dir ? gx_b   : gx_f;
    const int m0 = mT * 16, n0 = nT * 16;

    v8f acc = {};
#pragma unroll 4
    for (int k0 = 0; k0 < K; k0 += 32) {
        v16h a = load_a_frag(inp, K, m0, k0, lane);
        v16h b = load_b_frag(w,   K, n0, k0, lane);
        acc = wmma_f16(a, b, acc);
    }
    const int col   = n0 + (lane & 15);
    const int rbase = m0 + (lane >> 4) * 8;
    const float bv  = bias[col];
#pragma unroll
    for (int r = 0; r < 8; ++r)
        gx[(size_t)(rbase + r) * NG + col] = (_Float16)(acc[r] + bv);
}

// --- recurrent GEMM: gates = gx[t] + h @ Whh^T -----------------------------
// grid (16, 1, 2), block 256 (8 waves; wave -> nTile, register-blocks 4 M
// tiles). h staged to LDS via TDM (tensor_load_to_lds) or cooperative copy.
__global__ __launch_bounds__(256) void rec_gemm_kernel(
    const _Float16* __restrict__ h16,                        // [2][B*H]
    const _Float16* __restrict__ whh_f, const _Float16* __restrict__ whh_b,
    const _Float16* __restrict__ gx_f,  const _Float16* __restrict__ gx_b,
    float* __restrict__ gates,                               // [2][B*4H]
    int s) {
    __shared__ __align__(16) _Float16 hsh[BB * HH];          // 64 KB
    const int lane = threadIdx.x & 31;
    const int wave = threadIdx.x >> 5;
    const int dir  = blockIdx.z;
    const _Float16* w    = dir ? whh_b : whh_f;
    const _Float16* hsrc = h16 + (size_t)dir * BB * HH;

#if USE_TDM
    // Stage h[64,512] f16 into LDS with the Tensor Data Mover (one wave
    // issues; TENSORcnt tracks completion; barrier publishes to the block).
    if (wave == 0) {
        const unsigned lds_off = (unsigned)(uintptr_t)&hsh[0];
        const unsigned long long ga = (unsigned long long)(uintptr_t)hsrc;
        // D# group 0: count=1 | lds_addr | global_addr[56:0] | type=2
        u32x4 g0 = { 1u,
                     lds_off,
                     (unsigned)(ga & 0xFFFFFFFFu),
                     (unsigned)((ga >> 32) & 0x1FFFFFFu) | (2u << 30) };
        // D# group 1: data_size=2B; tensor 512x64; tile 512x64; stride0=512
        i32x8 g1 = { (int)(1u << 16),            // wg_mask=0, data_size=1 (2B)
                     (int)((unsigned)HH << 16),  // tensor_dim0[15:0]=512
                     (int)((unsigned)BB << 16),  // dim0 hi=0 | tensor_dim1=64
                     (int)((unsigned)HH << 16),  // dim1 hi=0 | tile_dim0=512
                     (int)BB,                    // tile_dim1=64, tile_dim2=0
                     (int)HH,                    // tensor_dim0_stride=512
                     0, 0 };
        i32x4 g2 = { 0, 0, 0, 0 };
        i32x4 g3 = { 0, 0, 0, 0 };
#if __clang_major__ >= 23
        i32x8 g4 = { 0, 0, 0, 0, 0, 0, 0, 0 };
        __builtin_amdgcn_tensor_load_to_lds(g0, g1, g2, g3, g4, 0);
#else
        __builtin_amdgcn_tensor_load_to_lds(g0, g1, g2, g3, 0);
#endif
        __builtin_amdgcn_s_wait_tensorcnt(0);
    }
    __syncthreads();
#else
    for (int i = threadIdx.x; i < (BB * HH) / 8; i += 256)
        ((v8h*)hsh)[i] = ((const v8h*)hsrc)[i];
    __syncthreads();
#endif

    const int nT = blockIdx.x * 8 + wave;   // 0..127
    const int n0 = nT * 16;
    const int tz = dir ? (TT - 1 - s) : s;
    const _Float16* gx = (dir ? gx_b : gx_f) + (size_t)tz * BB * NG;
    float* g = gates + (size_t)dir * BB * NG;

    v8f acc[4] = {{}, {}, {}, {}};
#pragma unroll 4
    for (int k0 = 0; k0 < HH; k0 += 32) {
        const v16h b = load_b_frag(w, HH, n0, k0, lane);
#pragma unroll
        for (int mt = 0; mt < 4; ++mt) {
            const v16h a = load_a_frag(hsh, HH, mt * 16, k0, lane);  // ds_load
            acc[mt] = wmma_f16(a, b, acc[mt]);
        }
    }
    const int col = n0 + (lane & 15);
    const int rb  = (lane >> 4) * 8;
#pragma unroll
    for (int mt = 0; mt < 4; ++mt) {
#pragma unroll
        for (int r = 0; r < 8; ++r) {
            const size_t off = (size_t)(mt * 16 + rb + r) * NG + col;
            g[off] = acc[mt][r] + (float)gx[off];
        }
    }
}

// --- LSTM cell update (fwd+bwd via z) --------------------------------------
// grid (128, 1, 2), block 256 -> B*H = 32768 elems per dir
__global__ __launch_bounds__(256) void cell_kernel(
    const float* __restrict__ gates,        // [2][B*4H]
    const float* __restrict__ masks,        // [T,B]
    float* __restrict__ c32, float* __restrict__ h32,   // [2][B*H]
    _Float16* __restrict__ h16,                          // [2][B*H]
    _Float16* __restrict__ outbuf,                       // [T,B,2H]
    int s, int layer, int last,
    float* __restrict__ dout_hn, float* __restrict__ dout_cn) {
    const int dir = blockIdx.z;
    const int idx = blockIdx.x * blockDim.x + threadIdx.x;
    const int b   = idx >> 9;          // / 512
    const int h   = idx & (HH - 1);
    const int tz  = dir ? (TT - 1 - s) : s;
    const float* g = gates + (size_t)dir * BB * NG + (size_t)b * NG;
    const float iv = g[h], fv = g[HH + h], gv = g[2 * HH + h], ov = g[3 * HH + h];
    const float mt = masks[tz * BB + b];
    const size_t st = (size_t)dir * BB * HH + idx;
    const float c  = c32[st];
    const float cnv = (sigmoidf_(fv) * c + sigmoidf_(iv) * tanhf(gv)) * mt;
    const float hnv = sigmoidf_(ov) * tanhf(cnv) * mt;
    c32[st] = cnv;
    h32[st] = hnv;
    h16[st] = (_Float16)hnv;
    outbuf[((size_t)tz * BB + b) * D2H + dir * HH + h] = (_Float16)hnv;
    if (last) {
        const size_t o = (size_t)(layer * 2 + dir) * BB * HH + (size_t)b * HH + h;
        dout_hn[o] = hnv;
        dout_cn[o] = cnv;
    }
}

// --- highway: g = sigmoid(raw @ projW^T + pb); out = g*raw + (1-g)*inp -----
// grid (1024/16=64, 16384/(16*8)=128), block 256
__global__ __launch_bounds__(256) void highway_kernel(
    const _Float16* __restrict__ raw,     // [M,1024] bidir layer output
    const _Float16* __restrict__ projw,   // [1024,1024] f16
    const float* __restrict__ projb,      // [1024]
    _Float16* __restrict__ io,            // [M,1024] residual in / result out
    float* __restrict__ dout, int write_f32) {
    const int lane = threadIdx.x & 31;
    const int wave = threadIdx.x >> 5;
    const int nT   = blockIdx.x;
    const int mT   = blockIdx.y * 8 + wave;
    const int m0 = mT * 16, n0 = nT * 16;

    v8f acc = {};
#pragma unroll 4
    for (int k0 = 0; k0 < D2H; k0 += 32) {
        v16h a = load_a_frag(raw,   D2H, m0, k0, lane);
        v16h b = load_b_frag(projw, D2H, n0, k0, lane);
        acc = wmma_f16(a, b, acc);
    }
    const int col   = n0 + (lane & 15);
    const int rbase = m0 + (lane >> 4) * 8;
    const float bv  = projb[col];
#pragma unroll
    for (int r = 0; r < 8; ++r) {
        const size_t off = (size_t)(rbase + r) * D2H + col;
        const float gate = sigmoidf_(acc[r] + bv);
        const float res  = gate * (float)raw[off] + (1.0f - gate) * (float)io[off];
        io[off] = (_Float16)res;
        if (write_f32) dout[off] = res;
    }
}

// ---------------------------------------------------------------------------
extern "C" void kernel_launch(void* const* d_in, const int* in_sizes, int n_in,
                              void* d_out, int out_size, void* d_ws, size_t ws_size,
                              hipStream_t stream) {
    const float* x     = (const float*)d_in[0];
    const float* masks = (const float*)d_in[1];
    const float* fWih0 = (const float*)d_in[2];
    const float* fWihR = (const float*)d_in[3];
    const float* fWhh  = (const float*)d_in[4];
    const float* fb    = (const float*)d_in[5];
    const float* bWih0 = (const float*)d_in[6];
    const float* bWihR = (const float*)d_in[7];
    const float* bWhh  = (const float*)d_in[8];
    const float* bbias = (const float*)d_in[9];
    const float* fInit = (const float*)d_in[10];
    const float* bInit = (const float*)d_in[11];
    const float* projW = (const float*)d_in[12];
    const float* projB = (const float*)d_in[13];

    float* out32 = (float*)d_out;                       // [T,B,2H]
    float* hn    = out32 + (size_t)MROWS * D2H;         // [6,B,H]
    float* cnp   = hn + (size_t)6 * BB * HH;            // [6,B,H]

    // ---- workspace layout (f16-heavy) ----
    char*  base = (char*)d_ws;
    size_t off  = 0;
    auto alloc16 = [&](size_t elems) -> _Float16* {
        off = (off + 255) & ~(size_t)255;
        _Float16* p = (_Float16*)(base + off);
        off += elems * sizeof(_Float16);
        return p;
    };
    auto alloc32 = [&](size_t elems) -> float* {
        off = (off + 255) & ~(size_t)255;
        float* p = (float*)(base + off);
        off += elems * sizeof(float);
        return p;
    };

    _Float16* wih16f[3]; _Float16* wih16b[3];
    _Float16* whh16f[3]; _Float16* whh16b[3];
    _Float16* proj16[2];
    wih16f[0] = alloc16((size_t)NG * DIN0);
    wih16f[1] = alloc16((size_t)NG * D2H);
    wih16f[2] = alloc16((size_t)NG * D2H);
    wih16b[0] = alloc16((size_t)NG * DIN0);
    wih16b[1] = alloc16((size_t)NG * D2H);
    wih16b[2] = alloc16((size_t)NG * D2H);
    for (int l = 0; l < 3; ++l) { whh16f[l] = alloc16((size_t)NG * HH); }
    for (int l = 0; l < 3; ++l) { whh16b[l] = alloc16((size_t)NG * HH); }
    proj16[0] = alloc16((size_t)D2H * D2H);
    proj16[1] = alloc16((size_t)D2H * D2H);

    _Float16* x16  = alloc16((size_t)MROWS * DIN0);
    _Float16* buf1 = alloc16((size_t)MROWS * D2H);
    _Float16* buf2 = alloc16((size_t)MROWS * D2H);
    _Float16* gxf  = alloc16((size_t)MROWS * NG);
    _Float16* gxb  = alloc16((size_t)MROWS * NG);
    float*    g32  = alloc32((size_t)2 * BB * NG);
    float*    c32  = alloc32((size_t)2 * BB * HH);
    float*    h32  = alloc32((size_t)2 * BB * HH);
    _Float16* h16s = alloc16((size_t)2 * BB * HH);

    auto conv = [&](const float* s, _Float16* d, size_t n) {
        f32_to_f16_kernel<<<dim3((unsigned)((n + 255) / 256)), 256, 0, stream>>>(s, d, (int)n);
    };

    // ---- weight / input conversion to f16 ----
    conv(fWih0, wih16f[0], (size_t)NG * DIN0);
    conv(fWihR,                          wih16f[1], (size_t)NG * D2H);
    conv(fWihR + (size_t)NG * D2H,       wih16f[2], (size_t)NG * D2H);
    conv(bWih0, wih16b[0], (size_t)NG * DIN0);
    conv(bWihR,                          wih16b[1], (size_t)NG * D2H);
    conv(bWihR + (size_t)NG * D2H,       wih16b[2], (size_t)NG * D2H);
    for (int l = 0; l < 3; ++l) conv(fWhh + (size_t)l * NG * HH, whh16f[l], (size_t)NG * HH);
    for (int l = 0; l < 3; ++l) conv(bWhh + (size_t)l * NG * HH, whh16b[l], (size_t)NG * HH);
    conv(projW,                          proj16[0], (size_t)D2H * D2H);
    conv(projW + (size_t)D2H * D2H,      proj16[1], (size_t)D2H * D2H);
    conv(x, x16, (size_t)MROWS * DIN0);

    // ---- layers ----
    for (int layer = 0; layer < 3; ++layer) {
        const _Float16* inp = (layer == 0) ? x16 : buf1;
        const int K = (layer == 0) ? DIN0 : D2H;
        _Float16* rawout = (layer == 0) ? buf1 : buf2;

        init_state_kernel<<<dim3(128, 1, 2), 256, 0, stream>>>(
            fInit + (size_t)layer * 2 * HH, bInit + (size_t)layer * 2 * HH,
            c32, h32, h16s);

        gx_gemm_kernel<<<dim3(128, 128, 2), 256, 0, stream>>>(
            inp, K, wih16f[layer], wih16b[layer],
            fb + (size_t)layer * NG, bbias + (size_t)layer * NG, gxf, gxb);

        for (int s = 0; s < TT; ++s) {
            rec_gemm_kernel<<<dim3(16, 1, 2), 256, 0, stream>>>(
                h16s, whh16f[layer], whh16b[layer], gxf, gxb, g32, s);
            cell_kernel<<<dim3(128, 1, 2), 256, 0, stream>>>(
                g32, masks, c32, h32, h16s, rawout,
                s, layer, (s == TT - 1) ? 1 : 0, hn, cnp);
        }

        if (layer > 0) {
            highway_kernel<<<dim3(64, 128, 1), 256, 0, stream>>>(
                rawout, proj16[layer - 1], projB + (size_t)(layer - 1) * D2H,
                buf1, out32, (layer == 2) ? 1 : 0);
        }
    }
}
